// MixtralSparseMoeBlock_1967095021956
// MI455X (gfx1250) — matrix operand order repaired and checked by reading, not compiled, not run on hardware
//
#include <hip/hip_runtime.h>
#include <cstddef>

// ---------------------------------------------------------------------------
// Mixtral sparse MoE block for MI455X (gfx1250, wave32, WMMA).
// fp32 inputs -> bf16x3 split-precision WMMA (v_wmma_f32_16x16x32_bf16):
//   A = Ahi + Alo, B = Bhi + Blo (bf16 RNE + residual);
//   D += Ahi*Bhi + Ahi*Blo + Alo*Bhi   (Alo*Blo ~2^-16, dropped)
// Sparse routing: only top-2 experts per token are computed (4x less math
// than the reference's dense expert loop, identical result).
// Double-buffered LDS fragment staging: load(k+1) -> WMMA(k) -> store(k+1)
// -> one barrier per K-step, so HBM/L2 latency hides behind the WMMAs.
// Workspace layout (needs ~59 MB):
//   [0,32)                : int cnt[8]           (padded to 256 B)
//   [256, +E*T*4)         : int list[E][T]       gathered token ids
//   [.., +T*E*4)          : float wte[T][E]      renormalized gate weights
//   [.., +T*I*4)          : float G[T][I]        per-expert intermediate
// ---------------------------------------------------------------------------

typedef __attribute__((ext_vector_type(16))) __bf16 v16bf;
typedef __attribute__((ext_vector_type(8)))  float  v8f;

namespace {
constexpr int kT = 2048;   // tokens (BATCH*SEQ)
constexpr int kH = 2048;   // hidden
constexpr int kI = 7168;   // intermediate
constexpr int kE = 8;      // experts
constexpr int BM = 32;     // token tile
constexpr int BN = 64;     // output-column tile
constexpr int BK = 32;     // K step (one bf16 WMMA depth)
}

__device__ __forceinline__ unsigned short bf16_rne(float f) {
  unsigned u = __float_as_uint(f);
  u += 0x7FFFu + ((u >> 16) & 1u);           // round-to-nearest-even
  return (unsigned short)(u >> 16);
}
__device__ __forceinline__ float bf16_up(unsigned short h) {
  return __uint_as_float(((unsigned)h) << 16);
}
__device__ __forceinline__ void split2(float v, unsigned short& hi, unsigned short& lo) {
  hi = bf16_rne(v);
  lo = bf16_rne(v - bf16_up(hi));            // residual term
}

// ---------------------------------------------------------------------------
__global__ void moe_zero(float* __restrict__ out, int n, int* __restrict__ cnt) {
  int i = blockIdx.x * blockDim.x + threadIdx.x;
  if (i < n) out[i] = 0.0f;
  if (i < kE) cnt[i] = 0;
}

// ---------------------------------------------------------------------------
// Router: one block (128 threads) per token. fp32 logits, softmax, top-2,
// renormalize; append token to its experts' gather lists.
__global__ __launch_bounds__(128) void moe_router(
    const float* __restrict__ x, const float* __restrict__ gw,
    int* __restrict__ cnt, int* __restrict__ list, float* __restrict__ wte) {
  const int t = blockIdx.x;
  const int tid = threadIdx.x;

  float p[kE];
#pragma unroll
  for (int e = 0; e < kE; ++e) p[e] = 0.0f;

  const float* xr = x + (size_t)t * kH;
  for (int h = tid; h < kH; h += 128) {
    float xv = xr[h];
#pragma unroll
    for (int e = 0; e < kE; ++e) p[e] += xv * gw[e * kH + h];
  }

  __shared__ float red[kE][128];
#pragma unroll
  for (int e = 0; e < kE; ++e) red[e][tid] = p[e];
  __syncthreads();
  for (int s = 64; s > 0; s >>= 1) {
    if (tid < s) {
#pragma unroll
      for (int e = 0; e < kE; ++e) red[e][tid] += red[e][tid + s];
    }
    __syncthreads();
  }

  if (tid == 0) {
    float l[kE], pe[kE];
    float m = -1e30f;
#pragma unroll
    for (int e = 0; e < kE; ++e) { l[e] = red[e][0]; m = fmaxf(m, l[e]); }
#pragma unroll
    for (int e = 0; e < kE; ++e) pe[e] = __expf(l[e] - m);
    int i0 = 0;
#pragma unroll
    for (int e = 1; e < kE; ++e) if (pe[e] > pe[i0]) i0 = e;       // ties: lowest idx
    int i1 = (i0 == 0) ? 1 : 0;
#pragma unroll
    for (int e = 0; e < kE; ++e) if (e != i0 && pe[e] > pe[i1]) i1 = e;
    float denom = pe[i0] + pe[i1];
    float w0 = pe[i0] / denom, w1v = pe[i1] / denom;
    int p0 = atomicAdd(&cnt[i0], 1);
    list[i0 * kT + p0] = t;
    wte[t * kE + i0] = w0;
    int p1 = atomicAdd(&cnt[i1], 1);
    list[i1 * kT + p1] = t;
    wte[t * kE + i1] = w1v;
  }
}

// ---------------------------------------------------------------------------
// Phase A: G[slot, :] = silu(Xg @ w1[e]^T) * (Xg @ w3[e]^T) for gathered rows.
// 256 threads = 8 waves; block tile 32(m) x 64(n); K loop over H in steps of 32.
// LDS holds fragments pre-swizzled to the ISA bf16 16x16x32 A/B lane layouts so
// each lane's fragment load is one contiguous 32-byte ds read.
__global__ __launch_bounds__(256) void moe_gemm13(
    const float* __restrict__ x, const float* __restrict__ w1,
    const float* __restrict__ w3, const int* __restrict__ list,
    const int* __restrict__ cnt, float* __restrict__ G, int e) {
  const int ne = cnt[e];
  const int tileM = blockIdx.y;
  if (tileM * BM >= ne) return;            // block-uniform early exit
  const int tileN = blockIdx.x;

  // [buf][sub-tile][hi/lo][lane*16 + elem]
  __shared__ __align__(32) unsigned short sA[2][2][2][512];
  __shared__ __align__(32) unsigned short sB1[2][4][2][512];
  __shared__ __align__(32) unsigned short sB3[2][4][2][512];
  __shared__ int s_tok[BM];

  const int tid = threadIdx.x;
  if (tid < BM) {
    int slot = tileM * BM + tid;
    s_tok[tid] = (slot < ne) ? list[e * kT + slot] : 0;
  }
  __syncthreads();

  const int lane = tid & 31;
  const int wave = tid >> 5;
  const int wm = wave >> 2;   // 0..1 : m sub-tile
  const int wn = wave & 3;    // 0..3 : n sub-tile

  // staging geometry (fixed per thread)
  const int arow = tid >> 3, akk = (tid & 7) << 2;   // A: 4 floats/thread
  const int amsub = arow >> 4, aml = arow & 15;
  const int brow = tid >> 2, bkk = (tid & 3) << 3;   // B: 8 floats/thread/matrix
  const int bnsub = brow >> 4, bnl = brow & 15;

  const float* xrow  = x + (size_t)s_tok[arow] * kH + akk;
  const float* w1row = w1 + (size_t)e * kI * kH + (size_t)(tileN * BN + brow) * kH + bkk;
  const float* w3row = w3 + (size_t)e * kI * kH + (size_t)(tileN * BN + brow) * kH + bkk;

  float a_r[4], b1_r[8], b3_r[8];

  auto load_tile = [&](int k0) {
    const float4 av = *(const float4*)(xrow + k0);
    a_r[0] = av.x; a_r[1] = av.y; a_r[2] = av.z; a_r[3] = av.w;
#pragma unroll
    for (int q = 0; q < 8; ++q) { b1_r[q] = w1row[k0 + q]; b3_r[q] = w3row[k0 + q]; }
    if (k0 + 2 * BK < kH) {                 // gfx1250 global_prefetch path
      __builtin_prefetch(w1row + k0 + 2 * BK, 0, 0);
      __builtin_prefetch(w3row + k0 + 2 * BK, 0, 0);
    }
  };

  auto store_tile = [&](int buf) {
#pragma unroll
    for (int q = 0; q < 4; ++q) {
      int k = akk + q;
      int ln = aml + (((k >> 3) & 1) << 4);          // A lane map
      int j = ((k >> 4) << 3) | (k & 7);             // A element map
      unsigned short hi, lo;
      split2(a_r[q], hi, lo);
      sA[buf][amsub][0][ln * 16 + j] = hi;
      sA[buf][amsub][1][ln * 16 + j] = lo;
    }
#pragma unroll
    for (int q = 0; q < 8; ++q) {
      int k = bkk + q;
      int ln = bnl + (((k >> 3) & 1) << 4);          // B lane map
      int j = ((k >> 4) << 3) | (k & 7);             // B element map
      unsigned short hi, lo;
      split2(b1_r[q], hi, lo);
      sB1[buf][bnsub][0][ln * 16 + j] = hi;
      sB1[buf][bnsub][1][ln * 16 + j] = lo;
      split2(b3_r[q], hi, lo);
      sB3[buf][bnsub][0][ln * 16 + j] = hi;
      sB3[buf][bnsub][1][ln * 16 + j] = lo;
    }
  };

  v8f acc1 = {};
  v8f acc3 = {};

  load_tile(0);
  store_tile(0);
  __syncthreads();

  constexpr int NS = kH / BK;
  for (int step = 0; step < NS; ++step) {
    const int cur = step & 1;
    if (step + 1 < NS) load_tile((step + 1) * BK);   // overlap with WMMAs below

    const v16bf ah  = *(const v16bf*)&sA[cur][wm][0][lane << 4];
    const v16bf al  = *(const v16bf*)&sA[cur][wm][1][lane << 4];
    const v16bf b1h = *(const v16bf*)&sB1[cur][wn][0][lane << 4];
    const v16bf b1l = *(const v16bf*)&sB1[cur][wn][1][lane << 4];
    const v16bf b3h = *(const v16bf*)&sB3[cur][wn][0][lane << 4];
    const v16bf b3l = *(const v16bf*)&sB3[cur][wn][1][lane << 4];

    acc1 = __builtin_amdgcn_wmma_f32_16x16x32_bf16(false, ah, false, b1h, (short)0, acc1, false, false);
    acc1 = __builtin_amdgcn_wmma_f32_16x16x32_bf16(false, ah, false, b1l, (short)0, acc1, false, false);
    acc1 = __builtin_amdgcn_wmma_f32_16x16x32_bf16(false, al, false, b1h, (short)0, acc1, false, false);
    acc3 = __builtin_amdgcn_wmma_f32_16x16x32_bf16(false, ah, false, b3h, (short)0, acc3, false, false);
    acc3 = __builtin_amdgcn_wmma_f32_16x16x32_bf16(false, ah, false, b3l, (short)0, acc3, false, false);
    acc3 = __builtin_amdgcn_wmma_f32_16x16x32_bf16(false, al, false, b3h, (short)0, acc3, false, false);

    if (step + 1 < NS) store_tile(cur ^ 1);
    __syncthreads();
  }

  // epilogue: SwiGLU combine, scatter to G (compacted by slot)
  const int ncol = tileN * BN + wn * 16 + (lane & 15);
  const int mbase = tileM * BM + wm * 16 + ((lane >> 4) << 3);
#pragma unroll
  for (int r = 0; r < 8; ++r) {
    int slot = mbase + r;
    if (slot < ne) {
      float pv = acc1[r], qv = acc3[r];
      float sig = __builtin_amdgcn_rcpf(1.0f + __expf(-pv));   // v_rcp_f32, no IEEE div
      G[(size_t)slot * kI + ncol] = pv * sig * qv;             // silu(p) * q
    }
  }
}

// ---------------------------------------------------------------------------
// Phase B: out[token, :] += gate_weight * (G @ w2[e]^T). Same tiling; K = I.
__global__ __launch_bounds__(256) void moe_gemm2(
    const float* __restrict__ G, const float* __restrict__ w2,
    const int* __restrict__ list, const int* __restrict__ cnt,
    const float* __restrict__ wte, float* __restrict__ out, int e) {
  const int ne = cnt[e];
  const int tileM = blockIdx.y;
  if (tileM * BM >= ne) return;
  const int tileN = blockIdx.x;

  __shared__ __align__(32) unsigned short sA[2][2][2][512];
  __shared__ __align__(32) unsigned short sB[2][4][2][512];
  __shared__ int s_tok[BM];
  __shared__ float s_scl[BM];

  const int tid = threadIdx.x;
  if (tid < BM) {
    int slot = tileM * BM + tid;
    int tok = (slot < ne) ? list[e * kT + slot] : 0;
    s_tok[tid] = tok;
    s_scl[tid] = (slot < ne) ? wte[tok * kE + e] : 0.0f;
  }
  __syncthreads();

  const int lane = tid & 31;
  const int wave = tid >> 5;
  const int wm = wave >> 2;
  const int wn = wave & 3;

  const int arow = tid >> 3, akk = (tid & 7) << 2;
  const int amsub = arow >> 4, aml = arow & 15;
  const int brow = tid >> 2, bkk = (tid & 3) << 3;
  const int bnsub = brow >> 4, bnl = brow & 15;

  const float* grow  = G + (size_t)(tileM * BM + arow) * kI + akk;
  const float* w2row = w2 + (size_t)e * kH * kI + (size_t)(tileN * BN + brow) * kI + bkk;

  float a_r[4], b_r[8];

  auto load_tile = [&](int k0) {
    const float4 av = *(const float4*)(grow + k0);
    a_r[0] = av.x; a_r[1] = av.y; a_r[2] = av.z; a_r[3] = av.w;
#pragma unroll
    for (int q = 0; q < 8; ++q) b_r[q] = w2row[k0 + q];
    if (k0 + 2 * BK < kI) __builtin_prefetch(w2row + k0 + 2 * BK, 0, 0);
  };

  auto store_tile = [&](int buf) {
#pragma unroll
    for (int q = 0; q < 4; ++q) {
      int k = akk + q;
      int ln = aml + (((k >> 3) & 1) << 4);
      int j = ((k >> 4) << 3) | (k & 7);
      unsigned short hi, lo;
      split2(a_r[q], hi, lo);
      sA[buf][amsub][0][ln * 16 + j] = hi;
      sA[buf][amsub][1][ln * 16 + j] = lo;
    }
#pragma unroll
    for (int q = 0; q < 8; ++q) {
      int k = bkk + q;
      int ln = bnl + (((k >> 3) & 1) << 4);
      int j = ((k >> 4) << 3) | (k & 7);
      unsigned short hi, lo;
      split2(b_r[q], hi, lo);
      sB[buf][bnsub][0][ln * 16 + j] = hi;
      sB[buf][bnsub][1][ln * 16 + j] = lo;
    }
  };

  v8f acc = {};

  load_tile(0);
  store_tile(0);
  __syncthreads();

  constexpr int NS = kI / BK;
  for (int step = 0; step < NS; ++step) {
    const int cur = step & 1;
    if (step + 1 < NS) load_tile((step + 1) * BK);

    const v16bf ah = *(const v16bf*)&sA[cur][wm][0][lane << 4];
    const v16bf al = *(const v16bf*)&sA[cur][wm][1][lane << 4];
    const v16bf bh = *(const v16bf*)&sB[cur][wn][0][lane << 4];
    const v16bf bl = *(const v16bf*)&sB[cur][wn][1][lane << 4];

    acc = __builtin_amdgcn_wmma_f32_16x16x32_bf16(false, ah, false, bh, (short)0, acc, false, false);
    acc = __builtin_amdgcn_wmma_f32_16x16x32_bf16(false, ah, false, bl, (short)0, acc, false, false);
    acc = __builtin_amdgcn_wmma_f32_16x16x32_bf16(false, al, false, bh, (short)0, acc, false, false);

    if (step + 1 < NS) store_tile(cur ^ 1);
    __syncthreads();
  }

  // epilogue: scatter-add weighted rows into out. Experts are serialized on
  // the stream and each token appears once per expert list -> plain RMW is safe.
  const int ncol = tileN * BN + wn * 16 + (lane & 15);
  const int mloc_base = wm * 16 + ((lane >> 4) << 3);
#pragma unroll
  for (int r = 0; r < 8; ++r) {
    int mloc = mloc_base + r;
    int slot = tileM * BM + mloc;
    if (slot < ne) {
      int tok = s_tok[mloc];
      float v = s_scl[mloc] * acc[r];
      size_t oi = (size_t)tok * kH + ncol;
      out[oi] = out[oi] + v;
    }
  }
}

// ---------------------------------------------------------------------------
extern "C" void kernel_launch(void* const* d_in, const int* in_sizes, int n_in,
                              void* d_out, int out_size, void* d_ws, size_t ws_size,
                              hipStream_t stream) {
  (void)in_sizes; (void)n_in; (void)out_size; (void)ws_size;
  const float* x  = (const float*)d_in[0];   // [1, 2048, 2048]
  const float* gw = (const float*)d_in[1];   // [8, 2048]
  const float* w1 = (const float*)d_in[2];   // [8, 7168, 2048]
  const float* w2 = (const float*)d_in[3];   // [8, 2048, 7168]
  const float* w3 = (const float*)d_in[4];   // [8, 7168, 2048]
  float* out = (float*)d_out;                // [1, 2048, 2048] fp32

  char* ws = (char*)d_ws;
  int*   cnt  = (int*)ws;                                     // 8 ints (256 B pad)
  int*   list = (int*)(ws + 256);                             // E*T ints
  float* wte  = (float*)(ws + 256 + (size_t)kE * kT * 4);     // T*E floats
  float* G    = (float*)(ws + 256 + 2 * (size_t)kE * kT * 4); // T*I floats (~58.7 MB)

  moe_zero<<<(kT * kH + 255) / 256, 256, 0, stream>>>(out, kT * kH, cnt);
  moe_router<<<kT, 128, 0, stream>>>(x, gw, cnt, list, wte);

  for (int e = 0; e < kE; ++e) {
    moe_gemm13<<<dim3(kI / BN, kT / BM), 256, 0, stream>>>(x, w1, w3, list, cnt, G, e);
    moe_gemm2 <<<dim3(kH / BN, kT / BM), 256, 0, stream>>>(G, w2, list, cnt, wte, out, e);
  }
}
